// Attention_15882789060926
// MI455X (gfx1250) — matrix operand order repaired and checked by reading, not compiled, not run on hardware
//
#include <hip/hip_runtime.h>

#define H_HEADS 16
#define D_MODEL 1024
#define SEQ_N   2048
#define BATCH   4
#define HD      64

typedef __attribute__((ext_vector_type(16))) _Float16 v16h;
typedef __attribute__((ext_vector_type(8)))  float    v8f;

union FragU { v16h v; uint4 u4[2]; };
union H2   { _Float16 h[2]; unsigned int u; };

// ---------------------------------------------------------------------------
// Fragment loaders. Memory layout: row-major with the CONTRACTION dimension
// contiguous (stride = halves per row). Each fragment is two 16B loads.
// A (16x32 f16, ISA 7.12.2): lane<16 holds row=lane, K={kb..kb+7, 16+kb..16+kb+7}
// with kb = 8*(lane>=16); packed 2 halves per dword, K ascending in memory.
// ---------------------------------------------------------------------------
__device__ __forceinline__ v16h load_frag_a(const _Float16* base, int stride,
                                            int row0, int k0, int lane) {
  const int r  = lane & 15;
  const int kb = (lane >> 4) << 3;
  const _Float16* p = base + (size_t)(row0 + r) * stride + k0 + kb;
  FragU f;
  f.u4[0] = *(const uint4*)p;
  f.u4[1] = *(const uint4*)(p + 16);
  return f.v;
}

// B (32x16 f16): lane<16 holds col=lane with K={0..7,16..23}, lanes 16-31 same cols
// with K={8..15,24..31}; two 16B loads per lane (same K order as A).
__device__ __forceinline__ v16h load_frag_b(const _Float16* base, int stride,
                                            int n0, int k0, int lane) {
  const int n   = lane & 15;
  const int kcb = (lane >> 4) << 3;
  const _Float16* p = base + (size_t)(n0 + n) * stride + k0 + kcb;
  FragU f;
  f.u4[0] = *(const uint4*)p;
  f.u4[1] = *(const uint4*)(p + 16);
  return f.v;
}

__device__ __forceinline__ v8f wmma16(v16h a, v16h b, v8f c) {
  return __builtin_amdgcn_wmma_f32_16x16x32_f16(false, a, false, b, (short)0, c,
                                                false, false);
}

// ===========================================================================
// Kernel 1: qkv = x @ w_qkv + b_qkv  (M=8192, K=1024, N=3072)
// Software-pipelined staging; branchless scatter epilogue ->
// q[B,H,N,64] (pre-scaled by hd^-0.5), k[B,H,N,64], vt[B,H,64,N], all f16.
// ===========================================================================
__global__ __launch_bounds__(256) void qkv_gemm(const float* __restrict__ x,
                                                const float* __restrict__ w,
                                                const float* __restrict__ bias,
                                                _Float16* __restrict__ qw,
                                                _Float16* __restrict__ kw,
                                                _Float16* __restrict__ vtw) {
  __shared__ _Float16 ldsA[128 * 40];   // [m][k] f16, stride 40 halves
  __shared__ _Float16 ldsW[128 * 40];   // [n][k] f16 (transposed), stride 40
  const int tid  = threadIdx.x;
  const int lane = tid & 31, wave = tid >> 5;
  const int wm = wave >> 2, wn = wave & 3;      // 2x4 wave grid, 64x32 per wave
  const int n0 = blockIdx.x * 128;
  const int m0 = blockIdx.y * 128;

  const int arow = tid >> 3;          // 0..31 (x4 passes -> 128 rows)
  const int ac4  = (tid & 7) << 2;    // 0..28
  const int wkk  = tid >> 5;          // 0..7  (x4 passes -> 32 k rows)
  const int wc4  = (tid & 31) << 2;   // 0..124

  const v8f vz = {0.f, 0.f, 0.f, 0.f, 0.f, 0.f, 0.f, 0.f};
  v8f acc[4][2];
#pragma unroll
  for (int mt = 0; mt < 4; ++mt)
#pragma unroll
    for (int nt = 0; nt < 2; ++nt) acc[mt][nt] = vz;

  float4 xa[4], wa[4];
  // prologue global load (ks = 0)
#pragma unroll
  for (int p = 0; p < 4; ++p) {
    xa[p] = *(const float4*)(x + (size_t)(m0 + p * 32 + arow) * D_MODEL + ac4);
    wa[p] = *(const float4*)(w + (size_t)(p * 8 + wkk) * (3 * D_MODEL) + n0 + wc4);
  }

  for (int ks = 0; ks < D_MODEL; ks += 32) {
    // commit staged registers -> LDS (f16 convert; W transposed to [n][k])
#pragma unroll
    for (int p = 0; p < 4; ++p) {
      H2 lo, hi;
      lo.h[0] = (_Float16)xa[p].x; lo.h[1] = (_Float16)xa[p].y;
      hi.h[0] = (_Float16)xa[p].z; hi.h[1] = (_Float16)xa[p].w;
      *(unsigned int*)&ldsA[(p * 32 + arow) * 40 + ac4]     = lo.u;
      *(unsigned int*)&ldsA[(p * 32 + arow) * 40 + ac4 + 2] = hi.u;
      ldsW[(wc4 + 0) * 40 + p * 8 + wkk] = (_Float16)wa[p].x;
      ldsW[(wc4 + 1) * 40 + p * 8 + wkk] = (_Float16)wa[p].y;
      ldsW[(wc4 + 2) * 40 + p * 8 + wkk] = (_Float16)wa[p].z;
      ldsW[(wc4 + 3) * 40 + p * 8 + wkk] = (_Float16)wa[p].w;
    }
    __syncthreads();

    // issue next K-step's global loads; they overlap the ds_load+wmma below
    if (ks + 32 < D_MODEL) {
      const int kn = ks + 32;
#pragma unroll
      for (int p = 0; p < 4; ++p) {
        xa[p] = *(const float4*)(x + (size_t)(m0 + p * 32 + arow) * D_MODEL + kn + ac4);
        wa[p] = *(const float4*)(w + (size_t)(kn + p * 8 + wkk) * (3 * D_MODEL) + n0 + wc4);
      }
    }

    v16h af[4], bf[2];
#pragma unroll
    for (int mt = 0; mt < 4; ++mt)
      af[mt] = load_frag_a(ldsA, 40, wm * 64 + mt * 16, 0, lane);
#pragma unroll
    for (int nt = 0; nt < 2; ++nt)
      bf[nt] = load_frag_b(ldsW, 40, wn * 32 + nt * 16, 0, lane);
#pragma unroll
    for (int mt = 0; mt < 4; ++mt)
#pragma unroll
      for (int nt = 0; nt < 2; ++nt)
        acc[mt][nt] = wmma16(af[mt], bf[nt], acc[mt][nt]);
    __syncthreads();
  }

  // Branchless scatter epilogue. A 16-col tile never crosses a q/k/v or head
  // boundary; a 128-row tile never crosses a batch boundary (2048 % 128 == 0).
  const int colb = lane & 15;
  const int rb   = (lane >> 4) << 3;
  const int bb   = m0 >> 11;           // batch index, uniform per block
  const int nrow0 = m0 & 2047;         // seq position base within batch
#pragma unroll
  for (int nt = 0; nt < 2; ++nt) {
    const int c     = n0 + wn * 32 + nt * 16 + colb;
    const float bv  = bias[c];
    const int which = c >> 10;                 // 0=q 1=k 2=v
    const int rem   = c & 1023;
    const int hh    = rem >> 6, dd = rem & 63;
    const float mul = (which == 0) ? 0.125f : 1.0f;   // fold hd^-0.5 into Q
    _Float16* bp = (which == 0) ? qw : ((which == 1) ? kw : vtw);
    const size_t base_head = (size_t)(bb * H_HEADS + hh) * (SEQ_N * HD);
#pragma unroll
    for (int mt = 0; mt < 4; ++mt) {
      const int nbase = nrow0 + wm * 64 + mt * 16 + rb;
#pragma unroll
      for (int r = 0; r < 8; ++r) {
        const int nn = nbase + r;
        const size_t off_qk = (size_t)nn * HD + dd;       // [n][d]
        const size_t off_vt = (size_t)dd * SEQ_N + nn;    // [d][n] (V transposed)
        const size_t off = (which == 2) ? off_vt : off_qk;
        bp[base_head + off] = (_Float16)((acc[mt][nt][r] + bv) * mul);
      }
    }
  }
}

// ===========================================================================
// Kernel 2: flash attention. 1 wave per 16 query rows, 4 waves/block.
// Streams K/V in 32-row blocks; online softmax; never materializes attn.
// ===========================================================================
__global__ __launch_bounds__(128) void flash_attn(const _Float16* __restrict__ q,
                                                  const _Float16* __restrict__ k,
                                                  const _Float16* __restrict__ vt,
                                                  _Float16* __restrict__ ao) {
  __shared__ _Float16 plds[4][16 * 40];  // per-wave P staging (16x32, stride 40)
  const int tid = threadIdx.x, lane = tid & 31, wave = tid >> 5;
  const int bh = blockIdx.y;            // 0..63  (b*16+h)
  const int b  = bh >> 4, h = bh & 15;
  const int q0 = (blockIdx.x * 4 + wave) * 16;

  const _Float16* qb = q  + (size_t)bh * SEQ_N * HD;
  const _Float16* kb = k  + (size_t)bh * SEQ_N * HD;
  const _Float16* vb = vt + (size_t)bh * HD * SEQ_N;

  const v16h qf0 = load_frag_a(qb, HD, q0, 0, lane);   // Q rows, d 0..31
  const v16h qf1 = load_frag_a(qb, HD, q0, 32, lane);  // Q rows, d 32..63

  const v8f vz = {0.f, 0.f, 0.f, 0.f, 0.f, 0.f, 0.f, 0.f};
  v8f o[4];
#pragma unroll
  for (int nt = 0; nt < 4; ++nt) o[nt] = vz;
  float mrun[8], lrun[8];
#pragma unroll
  for (int r = 0; r < 8; ++r) { mrun[r] = -1.0e30f; lrun[r] = 0.f; }

  const int colb = lane & 15;
  const int rb   = (lane >> 4) << 3;
  _Float16* pl = plds[wave];

  for (int kv = 0; kv < SEQ_N; kv += 32) {
    if (kv + 32 < SEQ_N) {
      __builtin_prefetch(kb + (size_t)(kv + 32 + lane) * HD, 0, 0);
      __builtin_prefetch(vb + (size_t)(lane * 2) * SEQ_N + kv + 32, 0, 0);
      __builtin_prefetch(vb + (size_t)(lane * 2 + 1) * SEQ_N + kv + 32, 0, 0);
    }

    // S(16x32) = Q(16x64) . K^T : 4 WMMA (2 per 16-col half, K split 2x32)
    v8f s0 = vz, s1 = vz;
    s0 = wmma16(qf0, load_frag_b(kb, HD, kv,      0,  lane), s0);
    s0 = wmma16(qf1, load_frag_b(kb, HD, kv,      32, lane), s0);
    s1 = wmma16(qf0, load_frag_b(kb, HD, kv + 16, 0,  lane), s1);
    s1 = wmma16(qf1, load_frag_b(kb, HD, kv + 16, 32, lane), s1);

    // row max over the 32 keys (C layout: row r+8*(lane>=16), spread over 16 lanes)
    float mx[8];
#pragma unroll
    for (int r = 0; r < 8; ++r) mx[r] = fmaxf(s0[r], s1[r]);
#pragma unroll
    for (int msk = 8; msk >= 1; msk >>= 1)
#pragma unroll
      for (int r = 0; r < 8; ++r) mx[r] = fmaxf(mx[r], __shfl_xor(mx[r], msk, 32));

    float al[8];
#pragma unroll
    for (int r = 0; r < 8; ++r) {
      const float mn = fmaxf(mrun[r], mx[r]);
      al[r]   = __expf(mrun[r] - mn);
      mrun[r] = mn;
    }

    // P = exp(S - m), staged to LDS in A-fragment-friendly row-major form
    float sm[8];
#pragma unroll
    for (int r = 0; r < 8; ++r) {
      const float e0 = __expf(s0[r] - mrun[r]);
      const float e1 = __expf(s1[r] - mrun[r]);
      pl[(rb + r) * 40 + colb]      = (_Float16)e0;
      pl[(rb + r) * 40 + 16 + colb] = (_Float16)e1;
      sm[r] = e0 + e1;
    }
#pragma unroll
    for (int msk = 8; msk >= 1; msk >>= 1)
#pragma unroll
      for (int r = 0; r < 8; ++r) sm[r] += __shfl_xor(sm[r], msk, 32);

#pragma unroll
    for (int r = 0; r < 8; ++r) lrun[r] = lrun[r] * al[r] + sm[r];
#pragma unroll
    for (int nt = 0; nt < 4; ++nt)
#pragma unroll
      for (int r = 0; r < 8; ++r) o[nt][r] *= al[r];

    __syncthreads();  // order P stores -> fragment loads (DS waits)
    const v16h pf = load_frag_a(pl, 40, 0, 0, lane);
#pragma unroll
    for (int nt = 0; nt < 4; ++nt)
      o[nt] = wmma16(pf, load_frag_b(vb, SEQ_N, nt * 16, kv, lane), o[nt]);
    __syncthreads();
  }

  // normalize and store f16 to attn_out [B,N,1024]
  float inv[8];
#pragma unroll
  for (int r = 0; r < 8; ++r) inv[r] = 1.0f / lrun[r];
  _Float16* outp = ao + ((size_t)b * SEQ_N) * D_MODEL + (size_t)h * HD;
#pragma unroll
  for (int nt = 0; nt < 4; ++nt)
#pragma unroll
    for (int r = 0; r < 8; ++r) {
      const int M = q0 + rb + r;
      outp[(size_t)M * D_MODEL + nt * 16 + colb] = (_Float16)(o[nt][r] * inv[r]);
    }
}

// ===========================================================================
// Kernel 3: out = attn @ w_proj + b_proj  (M=8192, K=1024, N=1024), f32 out.
// A (f16 ws) fragments load straight from global; W transposed through LDS,
// software-pipelined.
// ===========================================================================
__global__ __launch_bounds__(256) void proj_gemm(const _Float16* __restrict__ a,
                                                 const float* __restrict__ w,
                                                 const float* __restrict__ bias,
                                                 float* __restrict__ out) {
  __shared__ _Float16 ldsW[128 * 40];
  const int tid  = threadIdx.x;
  const int lane = tid & 31, wave = tid >> 5;
  const int wm = wave >> 2, wn = wave & 3;
  const int n0 = blockIdx.x * 128;
  const int m0 = blockIdx.y * 128;

  const int wkk = tid >> 5;           // 0..7
  const int wc4 = (tid & 31) << 2;    // 0..124

  const v8f vz = {0.f, 0.f, 0.f, 0.f, 0.f, 0.f, 0.f, 0.f};
  v8f acc[4][2];
#pragma unroll
  for (int mt = 0; mt < 4; ++mt)
#pragma unroll
    for (int nt = 0; nt < 2; ++nt) acc[mt][nt] = vz;

  float4 wa[4];
#pragma unroll
  for (int p = 0; p < 4; ++p)
    wa[p] = *(const float4*)(w + (size_t)(p * 8 + wkk) * D_MODEL + n0 + wc4);

  for (int ks = 0; ks < D_MODEL; ks += 32) {
#pragma unroll
    for (int p = 0; p < 4; ++p) {
      ldsW[(wc4 + 0) * 40 + p * 8 + wkk] = (_Float16)wa[p].x;
      ldsW[(wc4 + 1) * 40 + p * 8 + wkk] = (_Float16)wa[p].y;
      ldsW[(wc4 + 2) * 40 + p * 8 + wkk] = (_Float16)wa[p].z;
      ldsW[(wc4 + 3) * 40 + p * 8 + wkk] = (_Float16)wa[p].w;
    }
    __syncthreads();

    if (ks + 32 < D_MODEL) {
      const int kn = ks + 32;
#pragma unroll
      for (int p = 0; p < 4; ++p)
        wa[p] = *(const float4*)(w + (size_t)(kn + p * 8 + wkk) * D_MODEL + n0 + wc4);
    }

    v16h af[4], bf[2];
#pragma unroll
    for (int mt = 0; mt < 4; ++mt)
      af[mt] = load_frag_a(a, D_MODEL, m0 + wm * 64 + mt * 16, ks, lane);
#pragma unroll
    for (int nt = 0; nt < 2; ++nt)
      bf[nt] = load_frag_b(ldsW, 40, wn * 32 + nt * 16, 0, lane);
#pragma unroll
    for (int mt = 0; mt < 4; ++mt)
#pragma unroll
      for (int nt = 0; nt < 2; ++nt)
        acc[mt][nt] = wmma16(af[mt], bf[nt], acc[mt][nt]);
    __syncthreads();
  }

  const int colb = lane & 15;
  const int rb   = (lane >> 4) << 3;
#pragma unroll
  for (int nt = 0; nt < 2; ++nt) {
    const int c    = n0 + wn * 32 + nt * 16 + colb;
    const float bv = bias[c];
#pragma unroll
    for (int mt = 0; mt < 4; ++mt) {
      const int mbase = m0 + wm * 64 + mt * 16 + rb;
#pragma unroll
      for (int r = 0; r < 8; ++r)
        out[(size_t)(mbase + r) * D_MODEL + c] = acc[mt][nt][r] + bv;
    }
  }
}

// ===========================================================================
extern "C" void kernel_launch(void* const* d_in, const int* in_sizes, int n_in,
                              void* d_out, int out_size, void* d_ws, size_t ws_size,
                              hipStream_t stream) {
  const float* x      = (const float*)d_in[0];
  const float* w_qkv  = (const float*)d_in[1];
  const float* b_qkv  = (const float*)d_in[2];
  const float* w_proj = (const float*)d_in[3];
  const float* b_proj = (const float*)d_in[4];
  float* out = (float*)d_out;

  const size_t per = (size_t)BATCH * H_HEADS * SEQ_N * HD;  // 8.39M f16 elems
  _Float16* qw = (_Float16*)d_ws;
  _Float16* kw = qw + per;
  _Float16* vw = kw + per;
  _Float16* aw = vw + per;  // attn output [B,N,1024] f16; total ws = 64 MiB

  qkv_gemm<<<dim3(3 * D_MODEL / 128, (BATCH * SEQ_N) / 128), 256, 0, stream>>>(
      x, w_qkv, b_qkv, qw, kw, vw);
  flash_attn<<<dim3(SEQ_N / 64, BATCH * H_HEADS), 128, 0, stream>>>(qw, kw, vw, aw);
  proj_gemm<<<dim3(D_MODEL / 128, (BATCH * SEQ_N) / 128), 256, 0, stream>>>(
      aw, w_proj, b_proj, out);
}
